// EfficientAttention_80341658239025
// MI455X (gfx1250) — matrix operand order repaired and checked by reading, not compiled, not run on hardware
//
#include <hip/hip_runtime.h>

typedef __attribute__((ext_vector_type(16))) _Float16 v16h;
typedef __attribute__((ext_vector_type(8)))  _Float16 v8h;
typedef __attribute__((ext_vector_type(8)))  float    v8f;
typedef __attribute__((ext_vector_type(4)))  float    v4f;

typedef __attribute__((address_space(1))) const void as1_void;
typedef __attribute__((address_space(3))) void       as3_void;

#if defined(__HIP_DEVICE_COMPILE__) && __has_builtin(__builtin_amdgcn_global_load_async_to_lds_b128)
#define USE_ASYNC 1
#else
#define USE_ASYNC 0
#endif

#define WMMA_F16(a, b, c) \
  __builtin_amdgcn_wmma_f32_16x16x32_f16(false, (a), false, (b), (short)0, (c), false, false)

__device__ __forceinline__ v16h cat8(v8h lo, v8h hi) {
  return __builtin_shufflevector(lo, hi, 0,1,2,3,4,5,6,7,8,9,10,11,12,13,14,15);
}

// 16-byte global -> LDS copy; async (ASYNCcnt) on CDNA5, sync fallback otherwise.
__device__ __forceinline__ void copy16(const _Float16* g, _Float16* l) {
#if USE_ASYNC
  typedef int v4i_vs __attribute__((__vector_size__(16)));
  typedef __attribute__((address_space(1))) v4i_vs* gp_t;
  typedef __attribute__((address_space(3))) v4i_vs* lp_t;
  __builtin_amdgcn_global_load_async_to_lds_b128((gp_t)(as1_void*)g, (lp_t)(as3_void*)l, 0, 0);
#else
  *(v8h*)l = *(const v8h*)g;
#endif
}

__device__ __forceinline__ void async_wait() {
#if USE_ASYNC
# if __has_builtin(__builtin_amdgcn_s_wait_asynccnt)
  __builtin_amdgcn_s_wait_asynccnt(0);
# else
  asm volatile("s_wait_asynccnt 0" ::: "memory");
# endif
#endif
}

// ---------------------------------------------------------------------------
// Prep: fp32 -> f16 bulk convert (8 elements / thread)
// ---------------------------------------------------------------------------
__global__ __launch_bounds__(256) void k_cvt(
    const float* __restrict__ src, _Float16* __restrict__ dst, int n8) {
  int i = blockIdx.x * 256 + threadIdx.x;
  if (i >= n8) return;
  const v4f* s = (const v4f*)(src + (size_t)i * 8);
  v4f a = s[0], b = s[1];
  v8h o;
  #pragma unroll
  for (int j = 0; j < 4; j++) { o[j] = (_Float16)a[j]; o[4 + j] = (_Float16)b[j]; }
  *(v8h*)(dst + (size_t)i * 8) = o;
}

// ---------------------------------------------------------------------------
// Prep: dst[r*Kd + k] = (f16) src[k*R + r]   (src is [Kd x R], dst is [R x Kd])
// ---------------------------------------------------------------------------
__global__ __launch_bounds__(256) void k_transpose_f16(
    const float* __restrict__ src, _Float16* __restrict__ dst, int R, int Kd) {
  int i = blockIdx.x * 256 + threadIdx.x;
  if (i >= R * Kd) return;
  int r = i / Kd, k = i - r * Kd;
  dst[i] = (_Float16)src[(size_t)k * R + r];
}

// ---------------------------------------------------------------------------
// GEMM1: qkv = Xh[32768x512](f16) @ WqkvT[1536x512]^T; epilogue elu+1 on q,k;
// scatter per-head f16 Q/K/V [bh][4096][64]. 8 waves (4Mx2N), tile 128x128,
// K-step 32, double-buffered async LDS staging.
// ---------------------------------------------------------------------------
__global__ __launch_bounds__(256) void k_gemm_qkv(
    const _Float16* __restrict__ Xh, const _Float16* __restrict__ Wt,
    _Float16* __restrict__ Q, _Float16* __restrict__ K, _Float16* __restrict__ V) {
  __shared__ _Float16 As[2][128 * 40];
  __shared__ _Float16 Bs[2][128 * 40];
  const int tid = threadIdx.x, lane = tid & 31, wid = tid >> 5;
  const int g = lane >> 4, r = lane & 15;
  const int wm = wid >> 1, wn = wid & 1;            // wave tile: 32(M) x 64(N)
  const int m0 = blockIdx.y * 128, n0 = blockIdx.x * 128;
  const _Float16* Ab = Xh + (size_t)m0 * 512;
  const _Float16* Bb = Wt + (size_t)n0 * 512;
  v8f acc[2][4] = {};

  auto stage = [&](int kk, int buf) {
    #pragma unroll
    for (int cc = 0; cc < 2; cc++) {
      int c = tid + cc * 256;                 // 512 chunks of 16B per operand
      int row = c >> 2, sub = (c & 3) * 8;    // 4 chunks of 8 halves per row
      copy16(Ab + (size_t)row * 512 + kk + sub, &As[buf][row * 40 + sub]);
      copy16(Bb + (size_t)row * 512 + kk + sub, &Bs[buf][row * 40 + sub]);
    }
  };

  stage(0, 0);
  async_wait();
  __syncthreads();

  for (int k0 = 0; k0 < 512; k0 += 32) {
    const int buf = (k0 >> 5) & 1;
    if (k0 + 32 < 512) stage(k0 + 32, buf ^ 1);   // prefetch next tile (async)
    v16h af[2], bf[4];
    #pragma unroll
    for (int i = 0; i < 2; i++) {
      const _Float16* pa = &As[buf][(wm * 32 + i * 16 + r) * 40 + 8 * g];
      af[i] = cat8(*(const v8h*)pa, *(const v8h*)(pa + 16));
    }
    #pragma unroll
    for (int j = 0; j < 4; j++)
      bf[j] = cat8(*(const v8h*)&Bs[buf][(wn * 64 + j * 16 + r) * 40 + 8 * g], *(const v8h*)&Bs[buf][(wn * 64 + j * 16 + r) * 40 + 8 * g + 16]);
    #pragma unroll
    for (int i = 0; i < 2; i++)
      #pragma unroll
      for (int j = 0; j < 4; j++)
        acc[i][j] = WMMA_F16(af[i], bf[j], acc[i][j]);
    async_wait();
    __syncthreads();
  }

  #pragma unroll
  for (int j = 0; j < 4; j++) {
    int n = n0 + wn * 64 + j * 16 + r;   // column in [0,1536)
    int which = n >> 9;                  // 0=q 1=k 2=v (uniform per tile)
    int cc = n & 511;
    int h = cc >> 6, e = cc & 63;
    _Float16* dst = (which == 0) ? Q : (which == 1) ? K : V;
    bool act = (which < 2);
    #pragma unroll
    for (int i = 0; i < 2; i++) {
      #pragma unroll
      for (int rr = 0; rr < 8; rr++) {
        int m = m0 + wm * 32 + i * 16 + rr + 8 * g;
        int b = m >> 12, t = m & 4095;
        float val = acc[i][j][rr];
        if (act) val = (val > 0.f) ? (val + 1.f) : __expf(val);
        dst[((size_t)(b * 8 + h) * 4096 + t) * 64 + e] = (_Float16)val;
      }
    }
  }
}

// ---------------------------------------------------------------------------
// ksum[bh][d] = sum_n K[bh][n][d]
// ---------------------------------------------------------------------------
__global__ __launch_bounds__(256) void k_ksum(
    const _Float16* __restrict__ Kf, float* __restrict__ ksum) {
  __shared__ float s[256];
  int bh = blockIdx.x;
  int d = threadIdx.x & 63, ch = threadIdx.x >> 6;
  const _Float16* p = Kf + ((size_t)bh * 4096 + ch * 1024) * 64 + d;
  float acc = 0.f;
  for (int n = 0; n < 1024; n++) acc += (float)p[(size_t)n * 64];
  s[threadIdx.x] = acc;
  __syncthreads();
  if (threadIdx.x < 64)
    ksum[bh * 64 + threadIdx.x] =
        s[threadIdx.x] + s[64 + threadIdx.x] + s[128 + threadIdx.x] + s[192 + threadIdx.x];
}

// ---------------------------------------------------------------------------
// kvT[bh][e][d] = sum_n V[bh][n][e] * K[bh][n][d]; 4 waves, double-buffered
// async 32-token tiles, transposed fragment gathers from LDS.
// ---------------------------------------------------------------------------
__global__ __launch_bounds__(128) void k_kv(
    const _Float16* __restrict__ Kf, const _Float16* __restrict__ Vf,
    _Float16* __restrict__ KVt) {
  __shared__ _Float16 ks[2][32 * 72];
  __shared__ _Float16 vs[2][32 * 72];
  int bh = blockIdx.x;
  int tid = threadIdx.x, lane = tid & 31, w = tid >> 5;
  int g = lane >> 4, r = lane & 15;
  int e0 = w * 16;
  const size_t base = (size_t)bh * 4096 * 64;
  v8f acc[4] = {};

  auto stage = [&](int nn, int buf) {
    #pragma unroll
    for (int cc = 0; cc < 2; cc++) {
      int c = tid + cc * 128;                 // 256 chunks of 16B per operand
      int row = c >> 3, sub = (c & 7) * 8;
      copy16(Kf + base + (size_t)(nn + row) * 64 + sub, &ks[buf][row * 72 + sub]);
      copy16(Vf + base + (size_t)(nn + row) * 64 + sub, &vs[buf][row * 72 + sub]);
    }
  };

  stage(0, 0);
  async_wait();
  __syncthreads();

  for (int n0 = 0; n0 < 4096; n0 += 32) {
    const int buf = (n0 >> 5) & 1;
    if (n0 + 32 < 4096) stage(n0 + 32, buf ^ 1);
    v16h a;
    #pragma unroll
    for (int h = 0; h < 16; h++) {
      int tok = 8 * g + ((h < 8) ? h : h + 8);       // A 16x32 f16 layout
      a[h] = vs[buf][tok * 72 + e0 + r];
    }
    #pragma unroll
    for (int j = 0; j < 4; j++) {
      v16h bfr;
      #pragma unroll
      for (int h = 0; h < 16; h++)
        bfr[h] = ks[buf][(8 * g + ((h < 8) ? h : h + 8)) * 72 + j * 16 + r]; // B 32x16 f16 layout
      acc[j] = WMMA_F16(a, bfr, acc[j]);
    }
    async_wait();
    __syncthreads();
  }
  #pragma unroll
  for (int j = 0; j < 4; j++)
    #pragma unroll
    for (int rr = 0; rr < 8; rr++) {
      int e = e0 + rr + 8 * g, d = j * 16 + r;
      KVt[((size_t)bh * 64 + e) * 64 + d] = (_Float16)acc[j][rr];
    }
}

// ---------------------------------------------------------------------------
// z[bh][n] = 1 / (q . ksum + 1e-6)
// ---------------------------------------------------------------------------
__global__ __launch_bounds__(256) void k_z(
    const _Float16* __restrict__ Qf, const float* __restrict__ ksum,
    float* __restrict__ Z) {
  __shared__ float ks[64];
  int i = blockIdx.x * 256 + threadIdx.x;    // bh*4096 + n
  int bh = i >> 12;
  if (threadIdx.x < 64) ks[threadIdx.x] = ksum[bh * 64 + threadIdx.x];
  __syncthreads();
  const _Float16* q = Qf + (size_t)i * 64;
  float acc = 0.f;
  #pragma unroll
  for (int d = 0; d < 64; d++) acc += (float)q[d] * ks[d];
  Z[i] = 1.f / (acc + 1e-6f);
}

// ---------------------------------------------------------------------------
// attn = (q @ kv) * z, written merged-head f16 [32768 x 512]; kvT in LDS.
// ---------------------------------------------------------------------------
__global__ __launch_bounds__(256) void k_attn(
    const _Float16* __restrict__ Qf, const _Float16* __restrict__ KVt,
    const float* __restrict__ Z, _Float16* __restrict__ A) {
  __shared__ _Float16 Bs[64 * 72];
  int bh = blockIdx.y, mchunk = blockIdx.x;
  int tid = threadIdx.x, lane = tid & 31, w = tid >> 5;
  int g = lane >> 4, r = lane & 15;
  {
    #pragma unroll
    for (int cc = 0; cc < 2; cc++) {
      int c = tid + cc * 256;                 // 512 chunks of 16B
      int row = c >> 3, sub = (c & 7) * 8;
      copy16(KVt + ((size_t)bh * 64 + row) * 64 + sub, &Bs[row * 72 + sub]);
    }
  }
  async_wait();
  __syncthreads();
  int m0 = mchunk * 128 + w * 16;
  const _Float16* qbase = Qf + ((size_t)bh * 4096 + m0 + r) * 64;
  v8f acc[4] = {};
  #pragma unroll
  for (int k0 = 0; k0 < 64; k0 += 32) {
    const _Float16* pa = qbase + k0 + 8 * g;
    v16h a = cat8(*(const v8h*)pa, *(const v8h*)(pa + 16));
    #pragma unroll
    for (int j = 0; j < 4; j++) {
      v16h bfr = cat8(*(const v8h*)&Bs[(j * 16 + r) * 72 + k0 + 8 * g], *(const v8h*)&Bs[(j * 16 + r) * 72 + k0 + 8 * g + 16]);
      acc[j] = WMMA_F16(a, bfr, acc[j]);
    }
  }
  int b = bh >> 3, h = bh & 7;
  #pragma unroll
  for (int rr = 0; rr < 8; rr++) {
    int m = m0 + rr + 8 * g;
    float zv = Z[(size_t)bh * 4096 + m];
    #pragma unroll
    for (int j = 0; j < 4; j++) {
      int e = j * 16 + r;
      A[((size_t)(b * 4096 + m)) * 512 + h * 64 + e] = (_Float16)(acc[j][rr] * zv);
    }
  }
}

// ---------------------------------------------------------------------------
// GEMM4: out = attn[32768x512](f16) @ WprojT[512x512]^T + bias, fp32 out.
// Same double-buffered async structure as GEMM1.
// ---------------------------------------------------------------------------
__global__ __launch_bounds__(256) void k_proj(
    const _Float16* __restrict__ Af, const _Float16* __restrict__ Wt,
    const float* __restrict__ bias, float* __restrict__ Out) {
  __shared__ _Float16 As[2][128 * 40];
  __shared__ _Float16 Bs[2][128 * 40];
  const int tid = threadIdx.x, lane = tid & 31, wid = tid >> 5;
  const int g = lane >> 4, r = lane & 15;
  const int wm = wid >> 1, wn = wid & 1;
  const int m0 = blockIdx.y * 128, n0 = blockIdx.x * 128;
  const _Float16* Ab = Af + (size_t)m0 * 512;
  const _Float16* Bb = Wt + (size_t)n0 * 512;
  v8f acc[2][4] = {};

  auto stage = [&](int kk, int buf) {
    #pragma unroll
    for (int cc = 0; cc < 2; cc++) {
      int c = tid + cc * 256;
      int row = c >> 2, sub = (c & 3) * 8;
      copy16(Ab + (size_t)row * 512 + kk + sub, &As[buf][row * 40 + sub]);
      copy16(Bb + (size_t)row * 512 + kk + sub, &Bs[buf][row * 40 + sub]);
    }
  };

  stage(0, 0);
  async_wait();
  __syncthreads();

  for (int k0 = 0; k0 < 512; k0 += 32) {
    const int buf = (k0 >> 5) & 1;
    if (k0 + 32 < 512) stage(k0 + 32, buf ^ 1);
    v16h af[2], bf[4];
    #pragma unroll
    for (int i = 0; i < 2; i++) {
      const _Float16* pa = &As[buf][(wm * 32 + i * 16 + r) * 40 + 8 * g];
      af[i] = cat8(*(const v8h*)pa, *(const v8h*)(pa + 16));
    }
    #pragma unroll
    for (int j = 0; j < 4; j++)
      bf[j] = cat8(*(const v8h*)&Bs[buf][(wn * 64 + j * 16 + r) * 40 + 8 * g], *(const v8h*)&Bs[buf][(wn * 64 + j * 16 + r) * 40 + 8 * g + 16]);
    #pragma unroll
    for (int i = 0; i < 2; i++)
      #pragma unroll
      for (int j = 0; j < 4; j++)
        acc[i][j] = WMMA_F16(af[i], bf[j], acc[i][j]);
    async_wait();
    __syncthreads();
  }

  #pragma unroll
  for (int j = 0; j < 4; j++) {
    int n = n0 + wn * 64 + j * 16 + r;
    float bv = bias[n];
    #pragma unroll
    for (int i = 0; i < 2; i++)
      #pragma unroll
      for (int rr = 0; rr < 8; rr++) {
        int m = m0 + wm * 32 + i * 16 + rr + 8 * g;
        Out[(size_t)m * 512 + n] = acc[i][j][rr] + bv;
      }
  }
}

// ---------------------------------------------------------------------------
extern "C" void kernel_launch(void* const* d_in, const int* in_sizes, int n_in,
                              void* d_out, int out_size, void* d_ws, size_t ws_size,
                              hipStream_t stream) {
  (void)in_sizes; (void)n_in; (void)out_size; (void)ws_size;
  const float* x      = (const float*)d_in[0];   // [8,64,64,512]
  const float* w_qkv  = (const float*)d_in[1];   // [512,1536]
  const float* w_proj = (const float*)d_in[2];   // [512,512]
  const float* b_proj = (const float*)d_in[3];   // [512]
  float* out = (float*)d_out;

  char* ws = (char*)d_ws;
  size_t off = 0;
  auto alloc = [&](size_t bytes) -> void* {
    void* p = ws + off;
    off += (bytes + 255) & ~(size_t)255;
    return p;
  };
  _Float16* Xh     = (_Float16*)alloc((size_t)32768 * 512 * 2);
  _Float16* wqkvT  = (_Float16*)alloc((size_t)1536 * 512 * 2);
  _Float16* wprojT = (_Float16*)alloc((size_t)512 * 512 * 2);
  _Float16* Q      = (_Float16*)alloc((size_t)64 * 4096 * 64 * 2);
  _Float16* K      = (_Float16*)alloc((size_t)64 * 4096 * 64 * 2);
  _Float16* V      = (_Float16*)alloc((size_t)64 * 4096 * 64 * 2);
  _Float16* KVt    = (_Float16*)alloc((size_t)64 * 64 * 64 * 2);
  float*    ksum   = (float*)alloc((size_t)64 * 64 * 4);
  float*    Z      = (float*)alloc((size_t)64 * 4096 * 4);
  _Float16* Att    = (_Float16*)alloc((size_t)32768 * 512 * 2);

  k_cvt<<<(32768 * 512 / 8 + 255) / 256, 256, 0, stream>>>(x, Xh, 32768 * 512 / 8);
  k_transpose_f16<<<(1536 * 512 + 255) / 256, 256, 0, stream>>>(w_qkv, wqkvT, 1536, 512);
  k_transpose_f16<<<(512 * 512 + 255) / 256, 256, 0, stream>>>(w_proj, wprojT, 512, 512);

  k_gemm_qkv<<<dim3(12, 256), 256, 0, stream>>>(Xh, wqkvT, Q, K, V);
  k_ksum<<<64, 256, 0, stream>>>(K, ksum);
  k_kv<<<64, 128, 0, stream>>>(K, V, KVt);
  k_z<<<1024, 256, 0, stream>>>(Q, ksum, Z);
  k_attn<<<dim3(32, 64), 256, 0, stream>>>(Q, KVt, Z, Att);
  k_proj<<<dim3(4, 256), 256, 0, stream>>>(Att, wprojT, b_proj, out);
}
